// UELM4_64020782514672
// MI455X (gfx1250) — compile-verified, matrix operand order checked by reading, not run on hardware
//
#include <hip/hip_runtime.h>
#include <math.h>

typedef __bf16 bf16;
typedef __attribute__((ext_vector_type(8)))  __bf16 v8bf;
typedef __attribute__((ext_vector_type(16))) __bf16 v16bf;
typedef __attribute__((ext_vector_type(8)))  float  v8f;
typedef __attribute__((ext_vector_type(4)))  unsigned int u32x4;
typedef __attribute__((ext_vector_type(8)))  int i32x8;
typedef __attribute__((ext_vector_type(4)))  int i32x4;

#define VOCAB 50257
#define DMODEL 256
#define KMEM 65536
#define NTOK 2048
#define SLK 64
#define BAND 8
#define NSLICE 8
#define TILE_BYTES 8192   /* 16 rows x 256 bf16 = one contiguous span of M_bf16 */

#if __has_builtin(__builtin_amdgcn_tensor_load_to_lds)
#define HAS_TDM 1
#endif

__device__ __forceinline__ v16bf load_frag(const bf16* p) {
    v8bf lo = *(const v8bf*)(p);
    v8bf hi = *(const v8bf*)(p + 16);
    return __builtin_shufflevector(lo, hi, 0,1,2,3,4,5,6,7,8,9,10,11,12,13,14,15);
}

__device__ __forceinline__ v8f wmma_bf16(v16bf a, v16bf b, v8f c) {
    return __builtin_amdgcn_wmma_f32_16x16x32_bf16(false, a, false, b, (short)0, c, false, false);
}

__device__ __forceinline__ void wait_tensor(int n) {
#if __has_builtin(__builtin_amdgcn_s_wait_tensorcnt)
    if (n == 0) __builtin_amdgcn_s_wait_tensorcnt(0);
    else        __builtin_amdgcn_s_wait_tensorcnt(1);
#else
    if (n == 0) asm volatile("s_wait_tensorcnt 0" ::: "memory");
    else        asm volatile("s_wait_tensorcnt 1" ::: "memory");
#endif
}

#ifdef HAS_TDM
// TDM: copy one contiguous 8KB B-tile (16 M-rows) from global to LDS.
// D# group0: count=1, lds_addr, 57-bit global addr, type=2.
// D# group1: data_size=3 (8B), tensor_dim0=1024, tensor_dim1=1, tile_dim0=1024,
//            tile_dim1=1, tensor_dim0_stride=1024.
__device__ __forceinline__ void tdm_issue(unsigned lds_addr, const bf16* gsrc) {
    unsigned long long ga = (unsigned long long)(size_t)gsrc;
    u32x4 g0;
    g0[0] = 1u;
    g0[1] = lds_addr;
    g0[2] = (unsigned)(ga & 0xFFFFFFFFu);
    g0[3] = (unsigned)((ga >> 32) & 0x01FFFFFFu) | 0x80000000u;   // type=2
    i32x8 g1;
    g1[0] = 0x00030000;   // data_size=8B
    g1[1] = 0x04000000;   // tensor_dim0[15:0]=1024 at bits[31:16]
    g1[2] = 0x00010000;   // tensor_dim0 hi=0, tensor_dim1=1
    g1[3] = 0x04000000;   // tile_dim0=1024
    g1[4] = 1;            // tile_dim1=1
    g1[5] = 1024;         // tensor_dim0_stride
    g1[6] = 0; g1[7] = 0;
    i32x4 gz4 = {0, 0, 0, 0};
    i32x8 gz8 = {0, 0, 0, 0, 0, 0, 0, 0};
    __builtin_amdgcn_tensor_load_to_lds(g0, g1, gz4, gz4, gz8, 0);
}
#else
__device__ __forceinline__ void stage_fallback(bf16* dst, const bf16* src, int tid) {
    const v8bf* s = (const v8bf*)src;
    v8bf* d = (v8bf*)dst;
    #pragma unroll
    for (int j = 0; j < 4; ++j) d[tid * 4 + j] = s[tid * 4 + j];   // 128 thr x 64B = 8KB
}
#endif

// ---------------- f32 -> bf16 bulk convert ----------------
__global__ void k_cvt_bf16(const float* __restrict__ src, bf16* __restrict__ dst, int n) {
    for (int i = blockIdx.x * blockDim.x + threadIdx.x; i < n; i += gridDim.x * blockDim.x)
        dst[i] = (bf16)src[i];
}

// ---------------- banded field Ab = A * band_mask ----------------
__global__ void k_band(const float* __restrict__ A, float* __restrict__ Ab) {
    int i = blockIdx.x, j = threadIdx.x;
    int d = i - j; if (d < 0) d = -d;
    Ab[i * DMODEL + j] = (d <= BAND) ? A[i * DMODEL + j] : 0.0f;
}

// ---------------- embedding gather + causal mean ----------------
__global__ __launch_bounds__(DMODEL) void k_embed(const int* __restrict__ tokens,
                                                  const float* __restrict__ ET,
                                                  bf16* __restrict__ Ebf,
                                                  float* __restrict__ X) {
    int d = threadIdx.x;
    float s = 0.0f;
    for (int i = 0; i < NTOK; ++i) {
        int tok = tokens[i];
        float e = ET[(size_t)tok * DMODEL + d];
        s += e;
        X[(size_t)i * DMODEL + d] = s / (float)(i + 1);
        Ebf[(size_t)i * DMODEL + d] = (bf16)e;
    }
}

// ---------------- fused scores GEMM + per-(row,slice) top-64 ----------------
// grid (32 row-groups, 8 column slices), 4 waves/block; each wave owns 16 rows,
// all waves share the TDM-staged B tile (double buffered in LDS).
__global__ __launch_bounds__(128) void k_scores_topk(const bf16* __restrict__ Ebf,
                                                     const bf16* __restrict__ Mbf,
                                                     float* __restrict__ pS,
                                                     int* __restrict__ pI) {
    __shared__ __align__(16) bf16 Btile[2][16 * DMODEL];   // 2 x 8KB
    __shared__ float tileS[4][16 * 16];
    __shared__ float topS[4][16][SLK];
    __shared__ int   topI[4][16][SLK];

    const int tid = threadIdx.x;
    const int w = tid >> 5, lane = tid & 31;
    const int nr = lane & 15, half = lane >> 4;
    const int row0 = blockIdx.x * 64 + w * 16;
    const int tbeg = blockIdx.y * (KMEM / 16 / NSLICE);    // 512 tiles per slice
    const int niter = KMEM / 16 / NSLICE;

    for (int j = lane; j < 16 * SLK; j += 32) {
        topS[w][j >> 6][j & 63] = -INFINITY;
        topI[w][j >> 6][j & 63] = 0;
    }

    v16bf afrag[8];
    const bf16* arow = Ebf + (size_t)(row0 + nr) * DMODEL + half * 8;
    #pragma unroll
    for (int kc = 0; kc < 8; ++kc) afrag[kc] = load_frag(arow + kc * 32);

    float curMin = -INFINITY; int minPos = 0;

#ifdef HAS_TDM
    const unsigned ldsBase = (unsigned)(size_t)&Btile[0][0];
    if (tid == 0) tdm_issue(ldsBase, Mbf + (size_t)(tbeg * 16) * DMODEL);
#endif
    __syncthreads();

    for (int i = 0; i < niter; ++i) {
        const int buf = i & 1;
        const int col0 = (tbeg + i) * 16;
#ifdef HAS_TDM
        __syncthreads();   // previous compute done: other buffer free
        if (tid == 0) {
            if (i + 1 < niter) {
                tdm_issue(ldsBase + (unsigned)((buf ^ 1) * TILE_BYTES),
                          Mbf + (size_t)((tbeg + i + 1) * 16) * DMODEL);
                wait_tensor(1);
            } else {
                wait_tensor(0);
            }
        }
        __syncthreads();   // current buffer visible to all waves
#else
        __syncthreads();
        stage_fallback(&Btile[0][0], Mbf + (size_t)col0 * DMODEL, tid);
        __syncthreads();
#endif
        const bf16* bbase = &Btile[buf][0] + nr * DMODEL + half * 8;
        v16bf bfr[8];
        #pragma unroll
        for (int kc = 0; kc < 8; ++kc) bfr[kc] = load_frag(bbase + kc * 32);
        v8f acc = {};
        #pragma unroll
        for (int kc = 0; kc < 8; ++kc) acc = wmma_bf16(afrag[kc], bfr[kc], acc);

        #pragma unroll
        for (int v = 0; v < 8; ++v) tileS[w][(v + 8 * half) * 16 + nr] = acc[v];
        asm volatile("s_wait_dscnt 0" ::: "memory");
        __builtin_amdgcn_wave_barrier();

        if (lane < 16) {
            const int r = lane;
            for (int c = 0; c < 16; ++c) {
                float s = tileS[w][r * 16 + c];
                if (s > curMin) {
                    topS[w][r][minPos] = s;
                    topI[w][r][minPos] = col0 + c;
                    float m = topS[w][r][0]; int mp = 0;
                    for (int j = 1; j < SLK; ++j) {
                        float x = topS[w][r][j];
                        if (x < m) { m = x; mp = j; }
                    }
                    curMin = m; minPos = mp;
                }
            }
        }
        __builtin_amdgcn_wave_barrier();
    }

    __syncthreads();
    // Each wave's list IS the per-(row,slice) top-64: dump to workspace.
    for (int j = tid; j < 64 * SLK; j += 128) {
        int r = j >> 6, sel = j & 63;
        size_t o = ((size_t)(blockIdx.x * 64 + r) * NSLICE + blockIdx.y) * SLK + sel;
        pS[o] = topS[r >> 4][r & 15][sel];
        pI[o] = topI[r >> 4][r & 15][sel];
    }
}

// ---------------- merge 8 slice-lists per row -> global top-64 ----------------
__global__ __launch_bounds__(32) void k_topk_merge(const float* __restrict__ pS,
                                                   const int* __restrict__ pI,
                                                   int* __restrict__ Kset) {
    __shared__ float s[NSLICE * SLK];
    __shared__ int   si[NSLICE * SLK];
    const int row = blockIdx.x, lane = threadIdx.x;
    for (int j = lane; j < NSLICE * SLK; j += 32) {
        s[j]  = pS[(size_t)row * NSLICE * SLK + j];
        si[j] = pI[(size_t)row * NSLICE * SLK + j];
    }
    asm volatile("s_wait_dscnt 0" ::: "memory");
    __builtin_amdgcn_wave_barrier();
    for (int sel = 0; sel < SLK; ++sel) {
        float best = -INFINITY; int bpos = 0x7FFFFFFF;
        for (int j = lane; j < NSLICE * SLK; j += 32)
            if (s[j] > best) { best = s[j]; bpos = j; }
        #pragma unroll
        for (int off = 16; off > 0; off >>= 1) {
            float ob = __shfl_xor(best, off, 32);
            int   op = __shfl_xor(bpos, off, 32);
            if (ob > best || (ob == best && op < bpos)) { best = ob; bpos = op; }
        }
        if (lane == 0) {
            Kset[(size_t)row * SLK + sel] = si[bpos];
            s[bpos] = -INFINITY;
        }
        asm volatile("s_wait_dscnt 0" ::: "memory");
        __builtin_amdgcn_wave_barrier();
    }
}

// ---------------- per-row ADMM solver (T=4), gathered slots staged in LDS ----------------
__global__ __launch_bounds__(DMODEL) void k_solver(const bf16* __restrict__ Mbf,
                                                   const int* __restrict__ Kset,
                                                   const float* __restrict__ X,
                                                   const float* __restrict__ Ab,
                                                   bf16* __restrict__ Ybf) {
    __shared__ bf16  Mg[SLK][DMODEL];
    __shared__ float P[SLK];
    __shared__ float Ylds[DMODEL];
    __shared__ float Vlds[DMODEL];
    __shared__ float gP[SLK];
    __shared__ float red[2];

    const int row = blockIdx.x;
    const int t = threadIdx.x;
    const int* ks = Kset + (size_t)row * SLK;

    for (int k = 0; k < SLK; ++k) {
        int idx = ks[k];
        Mg[k][t] = Mbf[(size_t)idx * DMODEL + t];
    }
    if (t < SLK) P[t] = 1.0f / (float)SLK;
    __syncthreads();

    const float x = X[(size_t)row * DMODEL + t];
    float ymp = 0.0f;
    for (int k = 0; k < SLK; ++k) ymp += P[k] * (float)Mg[k][t];
    float y = ymp, lam = 0.0f;

    for (int it = 0; it < 4; ++it) {
        ymp = 0.0f;
        for (int k = 0; k < SLK; ++k) ymp += P[k] * (float)Mg[k][t];
        float r = y - ymp;
        lam += r;                                   // RHO = 1
        Ylds[t] = y; __syncthreads();
        float by = 0.0f;
        int j0 = (t - BAND < 0) ? 0 : t - BAND;
        int j1 = (t + BAND > DMODEL - 1) ? DMODEL - 1 : t + BAND;
        for (int j = j0; j <= j1; ++j) by += Ab[t * DMODEL + j] * Ylds[j];
        float gY = by - x + lam + r;
        y -= 0.1f * gY;                             // TAU = 0.1
        Vlds[t] = (y - ymp) + lam; __syncthreads();
        {
            int w = t >> 5, lane = t & 31;
            for (int i = 0; i < 8; ++i) {
                int k = w * 8 + i;
                float s = 0.0f;
                for (int d = lane; d < DMODEL; d += 32) s += Vlds[d] * (float)Mg[k][d];
                for (int off = 16; off > 0; off >>= 1) s += __shfl_down(s, off, 32);
                if (lane == 0) gP[k] = -s;
            }
        }
        __syncthreads();
        if (t < SLK) Vlds[t] = __logf(P[t] + 1e-20f) - 0.1f * gP[t];   // ETA = 0.1
        __syncthreads();
        if (t == 0) {
            float mx = Vlds[0];
            for (int k = 1; k < SLK; ++k) mx = fmaxf(mx, Vlds[k]);
            float sm = 0.0f;
            for (int k = 0; k < SLK; ++k) sm += __expf(Vlds[k] - mx);
            red[0] = mx; red[1] = sm;
        }
        __syncthreads();
        if (t < SLK) P[t] = __expf(Vlds[t] - red[0]) / red[1];
        __syncthreads();
    }
    Ybf[(size_t)row * DMODEL + t] = (bf16)y;
}

// ---------------- logits = Y @ M[:VOCAB]^T, TDM-staged B, NT stores ----------------
__global__ __launch_bounds__(128) void k_logits(const bf16* __restrict__ Ybf,
                                                const bf16* __restrict__ Mbf,
                                                float* __restrict__ out) {
    __shared__ __align__(16) bf16 Btile[2][16 * DMODEL];

    const int tid = threadIdx.x;
    const int w = tid >> 5, lane = tid & 31;
    const int nr = lane & 15, half = lane >> 4;
    const int row0 = blockIdx.x * 64 + w * 16;

    const int NT = (VOCAB + 15) / 16;                       // 3142
    const int TPB = (NT + gridDim.y - 1) / gridDim.y;
    const int tbeg = blockIdx.y * TPB;
    int tend = tbeg + TPB; if (tend > NT) tend = NT;
    const int niter = tend - tbeg;
    if (niter <= 0) return;

    v16bf afrag[8];
    const bf16* arow = Ybf + (size_t)(row0 + nr) * DMODEL + half * 8;
    #pragma unroll
    for (int kc = 0; kc < 8; ++kc) afrag[kc] = load_frag(arow + kc * 32);

#ifdef HAS_TDM
    const unsigned ldsBase = (unsigned)(size_t)&Btile[0][0];
    if (tid == 0) tdm_issue(ldsBase, Mbf + (size_t)(tbeg * 16) * DMODEL);
#endif
    __syncthreads();

    for (int i = 0; i < niter; ++i) {
        const int buf = i & 1;
        const int col0 = (tbeg + i) * 16;
#ifdef HAS_TDM
        __syncthreads();
        if (tid == 0) {
            if (i + 1 < niter) {
                tdm_issue(ldsBase + (unsigned)((buf ^ 1) * TILE_BYTES),
                          Mbf + (size_t)((tbeg + i + 1) * 16) * DMODEL);
                wait_tensor(1);
            } else {
                wait_tensor(0);
            }
        }
        __syncthreads();
#else
        __syncthreads();
        stage_fallback(&Btile[0][0], Mbf + (size_t)col0 * DMODEL, tid);
        __syncthreads();
#endif
        const bf16* bbase = &Btile[buf][0] + nr * DMODEL + half * 8;
        v16bf bfr[8];
        #pragma unroll
        for (int kc = 0; kc < 8; ++kc) bfr[kc] = load_frag(bbase + kc * 32);
        v8f acc = {};
        #pragma unroll
        for (int kc = 0; kc < 8; ++kc) acc = wmma_bf16(afrag[kc], bfr[kc], acc);

        const int col = col0 + nr;
        if (col < VOCAB) {
            #pragma unroll
            for (int v = 0; v < 8; ++v) {
                int rr = row0 + v + 8 * half;
                __builtin_nontemporal_store(acc[v], &out[(size_t)rr * VOCAB + col]);
            }
        }
    }
}

extern "C" void kernel_launch(void* const* d_in, const int* in_sizes, int n_in,
                              void* d_out, int out_size, void* d_ws, size_t ws_size,
                              hipStream_t stream) {
    const int*   tokens = (const int*)d_in[0];
    const float* ET     = (const float*)d_in[1];
    const float* M      = (const float*)d_in[2];
    const float* A      = (const float*)d_in[3];
    float* out = (float*)d_out;

    char* ws = (char*)d_ws;
    size_t o = 0;
    bf16*  Mbf  = (bf16*)(ws + o);  o += (size_t)KMEM * DMODEL * 2;          // 33.5 MB
    bf16*  Ebf  = (bf16*)(ws + o);  o += (size_t)NTOK * DMODEL * 2;          // 1 MB
    bf16*  Ybf  = (bf16*)(ws + o);  o += (size_t)NTOK * DMODEL * 2;          // 1 MB
    float* X    = (float*)(ws + o); o += (size_t)NTOK * DMODEL * 4;          // 2 MB
    float* Ab   = (float*)(ws + o); o += (size_t)DMODEL * DMODEL * 4;        // 256 KB
    int*   Kset = (int*)(ws + o);   o += (size_t)NTOK * SLK * 4;             // 512 KB
    float* pS   = (float*)(ws + o); o += (size_t)NTOK * NSLICE * SLK * 4;    // 4 MB
    int*   pI   = (int*)(ws + o);   o += (size_t)NTOK * NSLICE * SLK * 4;    // 4 MB

    k_cvt_bf16<<<4096, 256, 0, stream>>>(M, Mbf, KMEM * DMODEL);
    k_band<<<DMODEL, DMODEL, 0, stream>>>(A, Ab);
    k_embed<<<1, DMODEL, 0, stream>>>(tokens, ET, Ebf, X);
    k_scores_topk<<<dim3(NTOK / 64, NSLICE), 128, 0, stream>>>(Ebf, Mbf, pS, pI);
    k_topk_merge<<<NTOK, 32, 0, stream>>>(pS, pI, Kset);
    k_solver<<<NTOK, DMODEL, 0, stream>>>(Mbf, Kset, X, Ab, Ybf);
    k_logits<<<dim3(NTOK / 64, 16), 128, 0, stream>>>(Ybf, Mbf, out);
}
